// HydraScaleLM_8572754723193
// MI455X (gfx1250) — compile-verified, matrix operand order checked by reading, not run on hardware
//
#include <hip/hip_runtime.h>
#include <math.h>

// ---------------------------------------------------------------- constants
#define C_V    32000
#define C_D    1024
#define C_DEPTH 4
#define C_S    512
#define C_B    2
#define C_N    16
#define C_KW   4      // conv kernel width
#define C_R    64
#define C_MLP  4096
#define C_T    (C_B * C_S)   // 1024 rows total
#define C_PP   128           // padded p-projection width (>= R + 2N = 96)

// WMMA GEMM tiling: 256x128 block, 8 waves, 64x64 per wave (16 WMMA / k-step)
#define BM 256
#define BN 128
#define BK 32
#define KP 20          // padded row stride in uint pairs (40 bf16 = 80B rows)

typedef __attribute__((ext_vector_type(16))) __bf16 v16bf;
typedef __attribute__((ext_vector_type(8)))  float  v8f;
typedef __attribute__((ext_vector_type(4)))  float  f32x4;
typedef __attribute__((ext_vector_type(4)))  __bf16 bf16x4;
typedef __attribute__((ext_vector_type(4)))  unsigned ui32x4;
typedef __attribute__((ext_vector_type(8)))  int    i32x8;
typedef __attribute__((ext_vector_type(4)))  int    i32x4;

// Tensor Data Mover path (guarded: falls back to cooperative loads if the
// builtin is absent; host pass always takes the fallback branch).
#if defined(__AMDGCN__) && __has_builtin(__builtin_amdgcn_tensor_load_to_lds) && \
    __has_builtin(__builtin_amdgcn_s_wait_tensorcnt)
#define HYDRA_TDM 1
#else
#define HYDRA_TDM 0
#endif

// ---------------------------------------------------------------- helpers
__device__ __forceinline__ float siluf(float v) { return v / (1.0f + __expf(-v)); }

// Stage next K-tiles into registers: A 256x32 (8 b128/thread), W 128x32 (4).
__device__ __forceinline__ void tile_load_regs(const float* __restrict__ A, int m0, int lda,
                                               const float* __restrict__ W, int n0, int ldw,
                                               int kt, int tid, f32x4 ra[8], f32x4 rw[4]) {
#pragma unroll
  for (int i = 0; i < 8; ++i) {
    int id = tid + 256 * i;            // 0..2047
    int row = id >> 3, kq = (id & 7) * 4;
    ra[i] = *(const f32x4*)(A + (size_t)(m0 + row) * lda + kt + kq);
  }
#pragma unroll
  for (int i = 0; i < 4; ++i) {
    int id = tid + 256 * i;            // 0..1023
    int row = id >> 3, kq = (id & 7) * 4;
    rw[i] = *(const f32x4*)(W + (size_t)(n0 + row) * ldw + kt + kq);
  }
}

// Convert staged registers to bf16 and store to LDS (padded 80B rows).
__device__ __forceinline__ void tile_store_lds(__bf16* As, __bf16* Ws, int tid,
                                               const f32x4 ra[8], const f32x4 rw[4]) {
#pragma unroll
  for (int i = 0; i < 8; ++i) {
    int id = tid + 256 * i;
    int row = id >> 3, kq = (id & 7) * 4;
    *(bf16x4*)(As + row * (KP * 2) + kq) = __builtin_convertvector(ra[i], bf16x4);
  }
#pragma unroll
  for (int i = 0; i < 4; ++i) {
    int id = tid + 256 * i;
    int row = id >> 3, kq = (id & 7) * 4;
    *(bf16x4*)(Ws + row * (KP * 2) + kq) = __builtin_convertvector(rw[i], bf16x4);
  }
}

// Extract one 16x32 bf16 fragment row per lane from LDS.
// Lane (row=lane&15, half=lane>>4) holds K = {8h..8h+7, 16+8h..16+8h+7}.
__device__ __forceinline__ v16bf load_frag(const __bf16* lds, int row, int half) {
  union { unsigned u[8]; v16bf v; } f;
  const unsigned* b0 = (const unsigned*)lds + row * KP + half * 4;  // 16B aligned
  uint4 lo = *(const uint4*)(b0);
  uint4 hi = *(const uint4*)(b0 + 8);
  f.u[0] = lo.x; f.u[1] = lo.y; f.u[2] = lo.z; f.u[3] = lo.w;
  f.u[4] = hi.x; f.u[5] = hi.y; f.u[6] = hi.z; f.u[7] = hi.w;
  return f.v;
}

// ---------------------------------------------------------------- WMMA GEMM
// C[M,N] (+)= act( A[M,K] * W[N,K]^T + bias[N] )
// Preconditions: M % 256 == 0, N % 128 == 0, K % 32 == 0 (all padded).
// Software-pipelined; 64x64 wave tile -> 16 WMMA per 8 fragment loads.
// ACT: 0=none, 1=silu, 2=gelu(exact), 3=softplus
template <bool BIAS, int ACT, bool ACCUM>
__global__ __launch_bounds__(256) void hydra_gemm(const float* __restrict__ A, int lda,
                                                  const float* __restrict__ W, int ldw,
                                                  const float* __restrict__ bias,
                                                  float* __restrict__ C, int ldc,
                                                  int K) {
  __shared__ __align__(16) __bf16 As[BM * KP * 2];   // 20 KB
  __shared__ __align__(16) __bf16 Ws[BN * KP * 2];   // 10 KB

  const int tid  = threadIdx.x;
  const int lane = tid & 31;
  const int wave = tid >> 5;
  const int half = lane >> 4;
  const int lrow = lane & 15;
  const int m0 = blockIdx.y * BM;
  const int n0 = blockIdx.x * BN;
  const int wm = (wave & 3) * 64;    // 4 waves over M -> 64 rows each
  const int wn = (wave >> 2) * 64;   // 2 waves over N -> 64 cols each

  v8f acc[4][4];
#pragma unroll
  for (int i = 0; i < 4; ++i)
#pragma unroll
    for (int j = 0; j < 4; ++j)
#pragma unroll
      for (int e = 0; e < 8; ++e) acc[i][j][e] = 0.0f;

  f32x4 ra[8], rw[4];
  tile_load_regs(A, m0, lda, W, n0, ldw, 0, tid, ra, rw);

  for (int kt = 0; kt < K; kt += BK) {
    tile_store_lds(As, Ws, tid, ra, rw);      // waits on staged loads here
    __syncthreads();
    if (kt + BK < K) {                        // issue next tile; lands during WMMAs
      tile_load_regs(A, m0, lda, W, n0, ldw, kt + BK, tid, ra, rw);
      if (kt + 2 * BK < K) {                  // prefetch tile after next (global_prefetch_b8)
        __builtin_prefetch(A + (size_t)(m0 + tid) * lda + kt + 2 * BK, 0, 1);
        __builtin_prefetch(W + (size_t)(n0 + (tid & 127)) * ldw + kt + 2 * BK, 0, 1);
      }
    }

    v16bf af[4], wf[4];
#pragma unroll
    for (int i = 0; i < 4; ++i) af[i] = load_frag(As, wm + i * 16 + lrow, half);
#pragma unroll
    for (int j = 0; j < 4; ++j) wf[j] = load_frag(Ws, wn + j * 16 + lrow, half);

#pragma unroll
    for (int i = 0; i < 4; ++i)
#pragma unroll
      for (int j = 0; j < 4; ++j)
        acc[i][j] = __builtin_amdgcn_wmma_f32_16x16x32_bf16(
            false, af[i], false, wf[j], (short)0, acc[i][j], false, false);

    __syncthreads();
  }

  // Epilogue. C/D layout: M = v + 8*half, N = lane&15.
#pragma unroll
  for (int i = 0; i < 4; ++i) {
#pragma unroll
    for (int j = 0; j < 4; ++j) {
      int gn = n0 + wn + j * 16 + lrow;
      float bv = 0.0f;
      if constexpr (BIAS) bv = bias[gn];
#pragma unroll
      for (int v = 0; v < 8; ++v) {
        int gm = m0 + wm + i * 16 + half * 8 + v;
        float val = acc[i][j][v] + bv;
        if constexpr (ACT == 1) val = siluf(val);
        else if constexpr (ACT == 2) val = 0.5f * val * (1.0f + erff(val * 0.70710678f));
        else if constexpr (ACT == 3) val = (val > 20.f) ? val : log1pf(__expf(val));
        size_t idx = (size_t)gm * ldc + gn;
        if constexpr (ACCUM) C[idx] += val; else C[idx] = val;
      }
    }
  }
}

// ---------------------------------------------------------------- LayerNorm
__global__ __launch_bounds__(256) void hydra_ln(const float* __restrict__ x,
                                                const float* __restrict__ s,
                                                const float* __restrict__ b,
                                                float* __restrict__ out) {
  __shared__ float red[256];
  const int row = blockIdx.x;
  const float* xr = x + (size_t)row * C_D;
  float val[4];
  float sum = 0.f;
#pragma unroll
  for (int i = 0; i < 4; ++i) { val[i] = xr[threadIdx.x + i * 256]; sum += val[i]; }
  red[threadIdx.x] = sum; __syncthreads();
  for (int off = 128; off > 0; off >>= 1) {
    if (threadIdx.x < off) red[threadIdx.x] += red[threadIdx.x + off];
    __syncthreads();
  }
  float mean = red[0] / C_D; __syncthreads();
  float vs = 0.f;
#pragma unroll
  for (int i = 0; i < 4; ++i) { float d = val[i] - mean; vs += d * d; }
  red[threadIdx.x] = vs; __syncthreads();
  for (int off = 128; off > 0; off >>= 1) {
    if (threadIdx.x < off) red[threadIdx.x] += red[threadIdx.x + off];
    __syncthreads();
  }
  float rstd = rsqrtf(red[0] / C_D + 1e-5f);
  float* o = out + (size_t)row * C_D;
#pragma unroll
  for (int i = 0; i < 4; ++i) {
    int c = threadIdx.x + i * 256;
    o[c] = (val[i] - mean) * rstd * s[c] + b[c];
  }
}

// ---------------------------------------------------------------- small kernels
__global__ void hydra_temb_sin(const int* __restrict__ ts, float* __restrict__ temb0) {
  int gid = blockIdx.x * 256 + threadIdx.x;
  if (gid >= C_B * C_D) return;
  int b = gid >> 10, i = gid & (C_D - 1);
  const int half = C_D / 2;
  float tval = (float)ts[b];
  int k = (i < half) ? i : (i - half);
  float f = __expf(-(logf(10000.0f) / half) * (float)k);
  temb0[gid] = (i < half) ? sinf(tval * f) : cosf(tval * f);
}

template <int ACT>
__global__ void hydra_tiny_fc(const float* __restrict__ A, const float* __restrict__ W,
                              const float* __restrict__ bias, float* __restrict__ C,
                              int Krun, int Nrun, int total) {
  int gid = blockIdx.x * 256 + threadIdx.x;
  if (gid >= total) return;
  int b = gid / Nrun, j = gid - b * Nrun;
  const f32x4* a = (const f32x4*)(A + (size_t)b * Krun);
  const f32x4* w = (const f32x4*)(W + (size_t)j * Krun);
  float acc = 0.f;
  int k4 = Krun >> 2;
  for (int k = 0; k < k4; ++k) {
    f32x4 av = a[k], wv = w[k];
    acc += av.x * wv.x + av.y * wv.y + av.z * wv.z + av.w * wv.w;
  }
  acc += bias[j];
  if (ACT == 1) acc = siluf(acc);
  C[gid] = acc;
}

__global__ void hydra_embed(const int* __restrict__ ids, const float* __restrict__ tok,
                            const float* __restrict__ temb, float* __restrict__ x) {
  int gid = blockIdx.x * 256 + threadIdx.x;     // T*D threads
  int d = gid & (C_D - 1);
  int t = gid >> 10;
  int b = t >> 9;
  x[gid] = tok[(size_t)ids[t] * C_D + d] + temb[b * C_D + d];
}

// zero-pad pW (DEPTH,96,D) -> (DEPTH,128,D) so the p-projection GEMM is full-tile
__global__ void hydra_pad_pw(const float* __restrict__ pW, float* __restrict__ pWp) {
  int gid = blockIdx.x * 256 + threadIdx.x;     // DEPTH*128*D threads
  if (gid >= C_DEPTH * C_PP * C_D) return;
  int l   = gid / (C_PP * C_D);
  int rem = gid - l * C_PP * C_D;
  int row = rem >> 10;            // /C_D
  int col = rem & (C_D - 1);
  pWp[gid] = (row < 96) ? pW[((size_t)l * 96 + row) * C_D + col] : 0.0f;
}

__global__ void hydra_dwconv_silu(const float* __restrict__ xn, const float* __restrict__ cw,
                                  float* __restrict__ u) {
  int gid = blockIdx.x * 256 + threadIdx.x;     // T*D threads
  int d = gid & (C_D - 1);
  int t = gid >> 10;
  int b = t >> 9, si = t & (C_S - 1);
  float acc = 0.f;
#pragma unroll
  for (int j = 0; j < C_KW; ++j) {
    int ss = si - (C_KW - 1) + j;
    if (ss >= 0)
      acc += xn[((size_t)(b * C_S + ss)) * C_D + d] * cw[d * C_KW + j];
  }
  u[gid] = siluf(acc);
}

// Sequential SSM scan: one thread per (b, d); 16 recurrent states in registers.
// Whole Bp/Cp slab for this batch (S x 32 f32 = 64 KB) is preloaded into LDS
// once — via the Tensor Data Mover when available — so the recurrence loop is
// barrier-free.
__global__ __launch_bounds__(256) void hydra_scan(const float* __restrict__ dt,
                                                  const float* __restrict__ u,
                                                  const float* __restrict__ p,
                                                  const float* __restrict__ z,
                                                  const float* __restrict__ A_log,
                                                  const float* __restrict__ Dp,
                                                  float* __restrict__ y) {
  __shared__ __align__(16) float sBC[C_S * 2 * C_N];   // 64 KB (<< 320 KB/WGP)
  const int tid = threadIdx.x;
  int gid = blockIdx.x * 256 + tid;             // B*D threads; blocks don't straddle b
  int b = gid >> 10;
  int d = gid & (C_D - 1);

#if HYDRA_TDM
  if (tid < 32) {  // wave 0 issues one TDM DMA for the whole workgroup
    // D# group 0: count=1 | lds_addr | global_addr(57b) | type=2 (ISA 8.3)
    unsigned lds_addr  = (unsigned)(uintptr_t)&sBC[0];        // low 32b = LDS offset
    unsigned long long ga = (unsigned long long)(uintptr_t)(p + (size_t)b * C_S * C_PP + C_R);
    ui32x4 g0;
    g0[0] = 1u;                                   // count=1, user descriptor
    g0[1] = lds_addr;
    g0[2] = (unsigned)ga;                         // global_addr[31:0]
    g0[3] = (unsigned)((ga >> 32) & 0x1FFFFFFu) | 0x80000000u;  // addr[56:32] | type=2
    // D# group 1 (ISA 8.4): data_size=4B, tensor 32x512 (stride 128), tile 32x512
    i32x8 g1;
    g1[0] = 0x00020000;          // workgroup_mask=0, data_size=2 (4 bytes)
    g1[1] = (int)(32u << 16);    // tensor_dim0 = 32 (bits 79:48, low half)
    g1[2] = (int)(512u << 16);   // tensor_dim1 = 512 (bits 111:80, low half)
    g1[3] = (int)(32u << 16);    // tile_dim0 = 32 (bits 127:112)
    g1[4] = 512;                 // tile_dim1 = 512 (bits 143:128)
    g1[5] = 128;                 // tensor_dim0_stride = 128 (bits 207:160)
    g1[6] = 0;
    g1[7] = 0;
    i32x4 gz = {0, 0, 0, 0};     // groups 2/3 unused for 2D tiles
#if __clang_major__ >= 23
    i32x8 gz8 = {0, 0, 0, 0, 0, 0, 0, 0};
    __builtin_amdgcn_tensor_load_to_lds(g0, g1, gz, gz, gz8, 0);
#else
    __builtin_amdgcn_tensor_load_to_lds(g0, g1, gz, gz, 0);
#endif
    __builtin_amdgcn_s_wait_tensorcnt(0);
  }
#else
  for (int i4 = tid; i4 < (C_S * 2 * C_N) / 4; i4 += 256) {
    int t = i4 >> 3, c4 = (i4 & 7) * 4;
    ((f32x4*)sBC)[i4] =
        *(const f32x4*)(p + ((size_t)b * C_S + t) * C_PP + C_R + c4);
  }
#endif
  __syncthreads();

  float Av[C_N], h[C_N];
#pragma unroll
  for (int n = 0; n < C_N; ++n) { Av[n] = -__expf(A_log[d * C_N + n]); h[n] = 0.f; }
  float Dv = Dp[d];
  for (int t = 0; t < C_S; ++t) {
    size_t row = (size_t)b * C_S + t;
    const float* bc = &sBC[t * 2 * C_N];
    float dtv = dt[row * C_D + d];
    float uv  = u[row * C_D + d];
    float zv  = z[row * C_D + d];
    float acc = 0.f;
#pragma unroll
    for (int n = 0; n < C_N; ++n) {
      float a  = Av[n];
      float At = __expf(dtv * a);
      float Bt = (fabsf(a) < 1e-8f) ? dtv : (At - 1.0f) / (a + 1e-10f);
      h[n] = At * h[n] + Bt * bc[n] * uv;
      acc += bc[C_N + n] * h[n];
    }
    y[row * C_D + d] = (acc + uv * Dv) * siluf(zv);
  }
}

// ---------------------------------------------------------------- launch
extern "C" void kernel_launch(void* const* d_in, const int* in_sizes, int n_in,
                              void* d_out, int out_size, void* d_ws, size_t ws_size,
                              hipStream_t stream) {
  (void)in_sizes; (void)n_in; (void)out_size; (void)ws_size;

  const int*   input_ids = (const int*)  d_in[0];
  const int*   timesteps = (const int*)  d_in[1];
  const float* tok_emb   = (const float*)d_in[2];
  const float* time_w1   = (const float*)d_in[3];
  const float* time_b1   = (const float*)d_in[4];
  const float* time_w2   = (const float*)d_in[5];
  const float* time_b2   = (const float*)d_in[6];
  const float* norm1_s   = (const float*)d_in[7];
  const float* norm1_b   = (const float*)d_in[8];
  /* d_in[9] = xW : unused by the reference forward */
  const float* zW        = (const float*)d_in[10];
  const float* pW        = (const float*)d_in[11];
  const float* convW     = (const float*)d_in[12];
  const float* dtW       = (const float*)d_in[13];
  const float* dtb       = (const float*)d_in[14];
  const float* A_log     = (const float*)d_in[15];
  const float* Dp        = (const float*)d_in[16];
  const float* outW      = (const float*)d_in[17];
  const float* norm2_s   = (const float*)d_in[18];
  const float* norm2_b   = (const float*)d_in[19];
  const float* mlpW1     = (const float*)d_in[20];
  const float* mlpb1     = (const float*)d_in[21];
  const float* mlpW2     = (const float*)d_in[22];
  const float* mlpb2     = (const float*)d_in[23];
  const float* normo_s   = (const float*)d_in[24];
  const float* normo_b   = (const float*)d_in[25];
  const float* headW     = (const float*)d_in[26];
  const float* headb     = (const float*)d_in[27];
  float* out = (float*)d_out;

  // workspace layout (floats): ~27.8 MB total
  float* wsf = (float*)d_ws;
  const size_t TD = (size_t)C_T * C_D;
  float* x     = wsf;                        // T*D
  float* xn    = x + TD;                     // T*D
  float* scr   = xn + TD;                    // T*MLP  (= 4*T*D)
  float* zbuf  = scr;                        // phase 1: z
  float* dtbuf = scr + TD;                   //          dt
  float* ubuf  = scr + 2 * TD;               //          u
  float* ybuf  = scr + 3 * TD;               //          y
  float* mlph  = scr;                        // phase 2: MLP hidden (aliases z..y)
  float* pbuf  = scr + (size_t)C_T * C_MLP;  // T*128 (padded p)
  float* pWpad = pbuf + (size_t)C_T * C_PP;  // DEPTH*128*D (zero-padded pW)
  float* temb0 = pWpad + (size_t)C_DEPTH * C_PP * C_D;  // B*D
  float* h1    = temb0 + C_B * C_D;          // B*MLP
  float* temb  = h1 + (size_t)C_B * C_MLP;   // B*D

  const dim3 blk(256);
  const dim3 gSq(C_D / BN, C_T / BM);        // 8 x 4

  // --- one-time per call: pad pW, time embedding, token embedding
  hydra_pad_pw<<<(C_DEPTH * C_PP * C_D + 255) / 256, blk, 0, stream>>>(pW, pWpad);
  hydra_temb_sin<<<(C_B * C_D + 255) / 256, blk, 0, stream>>>(timesteps, temb0);
  hydra_tiny_fc<1><<<(C_B * C_MLP + 255) / 256, blk, 0, stream>>>(
      temb0, time_w1, time_b1, h1, C_D, C_MLP, C_B * C_MLP);
  hydra_tiny_fc<0><<<(C_B * C_D + 255) / 256, blk, 0, stream>>>(
      h1, time_w2, time_b2, temb, C_MLP, C_D, C_B * C_D);
  hydra_embed<<<(unsigned)(TD / 256), blk, 0, stream>>>(input_ids, tok_emb, temb, x);

  // --- blocks
  for (int l = 0; l < C_DEPTH; ++l) {
    const float* zW_l  = zW    + (size_t)l * C_D * C_D;
    const float* pWp_l = pWpad + (size_t)l * C_PP * C_D;
    const float* cw_l  = convW + (size_t)l * C_D * C_KW;
    const float* dtW_l = dtW   + (size_t)l * C_D * C_R;
    const float* dtb_l = dtb   + (size_t)l * C_D;
    const float* Al_l  = A_log + (size_t)l * C_D * C_N;
    const float* Dp_l  = Dp    + (size_t)l * C_D;
    const float* oW_l  = outW  + (size_t)l * C_D * C_D;
    const float* w1_l  = mlpW1 + (size_t)l * C_MLP * C_D;
    const float* b1_l  = mlpb1 + (size_t)l * C_MLP;
    const float* w2_l  = mlpW2 + (size_t)l * C_D * C_MLP;
    const float* b2_l  = mlpb2 + (size_t)l * C_D;

    // xn = LN1(x)
    hydra_ln<<<C_T, blk, 0, stream>>>(x, norm1_s + l * C_D, norm1_b + l * C_D, xn);
    // p = xn @ pWpad^T   (N padded to 128; cols 96..127 are zeros)
    hydra_gemm<false, 0, false><<<dim3(1, C_T / BM), blk, 0, stream>>>(
        xn, C_D, pWp_l, C_D, nullptr, pbuf, C_PP, C_D);
    // z = xn @ zW^T
    hydra_gemm<false, 0, false><<<gSq, blk, 0, stream>>>(
        xn, C_D, zW_l, C_D, nullptr, zbuf, C_D, C_D);
    // dt = softplus(p[:, :R] @ dtW^T + dtb)   (A strided: lda=128, K=64)
    hydra_gemm<true, 3, false><<<gSq, blk, 0, stream>>>(
        pbuf, C_PP, dtW_l, C_R, dtb_l, dtbuf, C_D, C_R);
    // u = silu(causal depthwise conv(xn))
    hydra_dwconv_silu<<<(unsigned)(TD / 256), blk, 0, stream>>>(xn, cw_l, ubuf);
    // selective scan -> y  (includes +u*Dp and *silu(z))
    hydra_scan<<<(C_B * C_D) / 256, blk, 0, stream>>>(
        dtbuf, ubuf, pbuf, zbuf, Al_l, Dp_l, ybuf);
    // x += y @ outW^T
    hydra_gemm<false, 0, true><<<gSq, blk, 0, stream>>>(
        ybuf, C_D, oW_l, C_D, nullptr, x, C_D, C_D);

    // MLP: xn = LN2(x);  x += gelu(xn@W1^T+b1) @ W2^T + b2
    hydra_ln<<<C_T, blk, 0, stream>>>(x, norm2_s + l * C_D, norm2_b + l * C_D, xn);
    hydra_gemm<true, 2, false><<<dim3(C_MLP / BN, C_T / BM), blk, 0, stream>>>(
        xn, C_D, w1_l, C_D, b1_l, mlph, C_MLP, C_D);
    hydra_gemm<true, 0, true><<<gSq, blk, 0, stream>>>(
        mlph, C_MLP, w2_l, C_MLP, b2_l, x, C_D, C_MLP);
  }

  // --- final LN + LM head (N = 32000 = 250 * 128, exact tiles)
  hydra_ln<<<C_T, blk, 0, stream>>>(x, normo_s, normo_b, xn);
  hydra_gemm<true, 0, false><<<dim3(C_V / BN, C_T / BM), blk, 0, stream>>>(
      xn, C_D, headW, C_D, headb, out, C_V, C_D);
}
